// Net_2989297238294
// MI455X (gfx1250) — matrix operand order repaired and checked by reading, not compiled, not run on hardware
//
#include <hip/hip_runtime.h>

typedef __attribute__((ext_vector_type(16))) _Float16 v16h;
typedef __attribute__((ext_vector_type(8)))  _Float16 v8h;
typedef __attribute__((ext_vector_type(8)))  float    v8f;
typedef __attribute__((ext_vector_type(4)))  unsigned int u32x4;
typedef __attribute__((ext_vector_type(8)))  int i32x8;
typedef __attribute__((ext_vector_type(4)))  int i32x4;

#define BATCH 4096
#define TSTEPS 128
#define DIN 78
#define HID 32
#define GATES 128   // 4*HID
#define KC 3        // K padded 78 -> 96 (col 78 carries the fused bias, x col 78 == 1.0)
#define NT 8        // 128/16 N tiles
#define WAVES 4
#define ROWS_PER_WAVE 16
#define ROWS_PER_BLOCK (WAVES * ROWS_PER_WAVE)   // 64
#define NBLOCKS (BATCH / ROWS_PER_BLOCK)         // 64
#define XCOLS 80    // LDS x tile row length (TDM zero-pads cols 78..79)

#if __has_builtin(__builtin_amdgcn_tensor_load_to_lds) && \
    __has_builtin(__builtin_amdgcn_s_wait_tensorcnt)
#define USE_TDM 1
#else
#define USE_TDM 0
#endif

union U16 { v16h v; v8h h2[2]; _Float16 e[16]; };
union U8f { v8f v; float e[8]; };

struct SharedMem {
  alignas(32) _Float16 fragWih[KC][NT][32][16];  // 24 KB, B-fragment packed
  alignas(32) _Float16 fragWhh[NT][32][16];      //  8 KB, B-fragment packed
  alignas(32) _Float16 hbuf[WAVES][16][HID];     //  4 KB, h transpose staging
#if USE_TDM
  alignas(32) float xtile[WAVES][16][XCOLS];     // 20 KB, TDM destination
#endif
};

// ---- Fast activations: gfx1250 v_tanh_f32 when available ----
#if __has_builtin(__builtin_amdgcn_tanhf)
__device__ __forceinline__ float tanh_fast(float x) {
  return __builtin_amdgcn_tanhf(x);
}
#else
__device__ __forceinline__ float tanh_fast(float x) {
  float a = fabsf(x);
  float e = __expf(-2.0f * a);
  float t = (1.0f - e) / (1.0f + e);
  return copysignf(t, x);
}
#endif
__device__ __forceinline__ float sigmoid_f(float x) {
  return __builtin_fmaf(0.5f, tanh_fast(0.5f * x), 0.5f);
}

#if USE_TDM
// One TDM op: copy a 16 x 78 f32 tile (rows strided TSTEPS*DIN) into LDS as
// 16 x 80 (cols 78..79 zero-filled by OOB-read semantics).
__device__ __forceinline__ void tdm_load_xtile(unsigned long long ga, unsigned lds) {
  u32x4 g0 = { 1u,                               // count=1 valid descriptor
               lds,                              // lds_addr
               (unsigned)ga,                     // global_addr[31:0]
               (unsigned)((ga >> 32) & 0x01FFFFFFu) | 0x80000000u }; // [56:32] | type=2
  i32x8 g1 = { (int)(2u << 16),                  // data_size=4B, wg_mask=0, no flags
               (int)((unsigned)DIN << 16),       // tensor_dim0 = 78 (low16)
               (int)(16u << 16),                 // tensor_dim1 = 16 (low16)
               (int)((unsigned)XCOLS << 16),     // tile_dim0 = 80
               16,                               // tile_dim1 = 16, tile_dim2 = 0
               (int)(TSTEPS * DIN),              // tensor_dim0_stride = 9984
               0, 0 };
  i32x4 gz = { 0, 0, 0, 0 };
#if defined(__clang_major__) && (__clang_major__ >= 23)
  i32x8 gz8 = { 0, 0, 0, 0, 0, 0, 0, 0 };
  __builtin_amdgcn_tensor_load_to_lds(g0, g1, gz, gz, gz8, 0);
#else
  __builtin_amdgcn_tensor_load_to_lds(g0, g1, gz, gz, 0);
#endif
}
#endif

// Convert 8 aligned LDS floats -> 8 f16 (2x ds_load_b128 + packed cvt).
__device__ __forceinline__ v8h cvtRun8(const float* p) {
  v8f v = *(const v8f*)p;
  return __builtin_convertvector(v, v8h);
}

#if !USE_TDM
// Fallback: global gather with zero padding; column DIN carries the bias 1.0.
__device__ __forceinline__ void loadRun8Global(const float* __restrict__ p, int d0,
                                               _Float16* dst) {
  if (d0 + 8 <= DIN) {
    float v[8];
    __builtin_memcpy(v, p + d0, 32);
#pragma unroll
    for (int i = 0; i < 8; ++i) dst[i] = (_Float16)v[i];
  } else {
#pragma unroll
    for (int i = 0; i < 8; ++i) {
      int d = d0 + i;
      float vv = (d < DIN) ? p[d] : (d == DIN ? 1.0f : 0.0f);
      dst[i] = (_Float16)vv;
    }
  }
}
#endif

__global__ __launch_bounds__(128) void lstm_fused_kernel(
    const float* __restrict__ x,    // [B, T, DIN]
    const float* __restrict__ Wih,  // [GATES, DIN]
    const float* __restrict__ Whh,  // [GATES, HID]
    const float* __restrict__ bih,  // [GATES]
    const float* __restrict__ bhh,  // [GATES]
    const float* __restrict__ Wout, // [2, HID]
    const float* __restrict__ bout, // [2]
    float* __restrict__ out)        // [B, 2]
{
  __shared__ SharedMem smem;
  const int tid = threadIdx.x;

  // ---- Stage weights into LDS as f16 WMMA B-fragments ----
  // B layout (KxN=32x16): lanes 0-15 -> n=lane, K=0..7,16..23 over 16 halves;
  // lanes 16-31 -> n=lane-16, K=8..15,24..31.  Row k==DIN holds the fused bias.
  for (int idx = tid; idx < KC * NT * 32 * 16; idx += 128) {
    int h0 = idx & 15;
    int l  = (idx >> 4) & 31;
    int nt = (idx >> 9) & 7;
    int kc = idx >> 12;
    int k = kc * 32 + (h0 & 8) * 2 + (l >> 4) * 8 + (h0 & 7);
    int n = nt * 16 + (l & 15);
    float v = (k < DIN) ? Wih[n * DIN + k]
                        : (k == DIN ? (bih[n] + bhh[n]) : 0.0f);
    smem.fragWih[kc][nt][l][h0] = (_Float16)v;
  }
  for (int idx = tid; idx < NT * 32 * 16; idx += 128) {
    int h0 = idx & 15;
    int l  = (idx >> 4) & 31;
    int nt = idx >> 9;
    int k = (h0 & 8) * 2 + (l >> 4) * 8 + (h0 & 7);
    int n = nt * 16 + (l & 15);
    smem.fragWhh[nt][l][h0] = (_Float16)Whh[n * HID + k];
  }
  __syncthreads();

  const int wave = tid >> 5;
  const int lane = tid & 31;
  const int lm   = lane & 15;   // A-row / C-column within tile
  const int lh   = lane >> 4;   // lane-half
  const int ko   = lh * 8;      // A-fragment K offset
  const int rowBase = blockIdx.x * ROWS_PER_BLOCK + wave * ROWS_PER_WAVE;

#if USE_TDM
  const int wave_u = __builtin_amdgcn_readfirstlane(wave);
  const unsigned ldsX = (unsigned)(__builtin_offsetof(SharedMem, xtile) +
                                   (size_t)wave_u * 16 * XCOLS * sizeof(float));
  const unsigned long long gbase =
      (unsigned long long)(uintptr_t)x +
      (unsigned long long)(blockIdx.x * ROWS_PER_BLOCK + wave_u * ROWS_PER_WAVE) *
          TSTEPS * DIN * sizeof(float);
  tdm_load_xtile(gbase, ldsX);                       // prologue: tile for t=0
#else
  const float* xrow = x + (size_t)(rowBase + lm) * TSTEPS * DIN;
#endif

  // Recurrent state: h as f16 A-fragment (16x32), c as two 16x16 f32 tiles.
  U16 hfrag;
#pragma unroll
  for (int i = 0; i < 16; ++i) hfrag.e[i] = (_Float16)0.0f;
  v8f c0 = {};
  v8f c1 = {};

  // ax2 high half (K=80..95) is constant zero: set once, keep live.
  U16 ax0, ax1, ax2;
#pragma unroll
  for (int i = 8; i < 16; ++i) ax2.e[i] = (_Float16)0.0f;

#pragma unroll 1
  for (int t = 0; t < TSTEPS; ++t) {
    // ---- x A-fragments (16x32 f16 each, K padded to 96, col 78 = 1.0) ----
#if USE_TDM
    __builtin_amdgcn_s_wait_tensorcnt(0);            // tile t landed in LDS
    if (lane < 16) smem.xtile[wave][lane][DIN] = 1.0f;  // bias column
    const float* xr = &smem.xtile[wave][lm][0];
    ax0.h2[0] = cvtRun8(xr + ko);
    ax0.h2[1] = cvtRun8(xr + ko + 16);
    ax1.h2[0] = cvtRun8(xr + 32 + ko);
    ax1.h2[1] = cvtRun8(xr + 32 + ko + 16);
    ax2.h2[0] = cvtRun8(xr + 64 + ko);               // cols 64..79 (incl bias)
    // All tile reads done; safe to overwrite with next DMA.
    asm volatile("s_wait_dscnt 0x0" ::: "memory");
    if (t + 1 < TSTEPS)
      tdm_load_xtile(gbase + (unsigned long long)(t + 1) * DIN * sizeof(float), ldsX);
#else
    const float* xp = xrow + (size_t)t * DIN;
    loadRun8Global(xp, ko,           &ax0.e[0]);
    loadRun8Global(xp, ko + 16,      &ax0.e[8]);
    loadRun8Global(xp, 32 + ko,      &ax1.e[0]);
    loadRun8Global(xp, 32 + ko + 16, &ax1.e[8]);
    loadRun8Global(xp, 64 + ko,      &ax2.e[0]);
    loadRun8Global(xp, 64 + ko + 16, &ax2.e[8]);
    if (t + 1 < TSTEPS) __builtin_prefetch(xp + DIN, 0, 0);
#endif

    // ---- GEMMs with double-buffered B-fragment prefetch ----
    // Chain per tile: kc2 (bias, C=inline 0) -> kc0 -> kc1 -> hidden.
    v16h bs[2][4];
    bs[0][0] = *(const v16h*)&smem.fragWih[2][0][lane][0];
    bs[0][1] = *(const v16h*)&smem.fragWih[0][0][lane][0];
    bs[0][2] = *(const v16h*)&smem.fragWih[1][0][lane][0];
    bs[0][3] = *(const v16h*)&smem.fragWhh[0][lane][0];
    v8f acc[NT];
#pragma unroll
    for (int nt = 0; nt < NT; ++nt) {
      if (nt + 1 < NT) {
        bs[(nt + 1) & 1][0] = *(const v16h*)&smem.fragWih[2][nt + 1][lane][0];
        bs[(nt + 1) & 1][1] = *(const v16h*)&smem.fragWih[0][nt + 1][lane][0];
        bs[(nt + 1) & 1][2] = *(const v16h*)&smem.fragWih[1][nt + 1][lane][0];
        bs[(nt + 1) & 1][3] = *(const v16h*)&smem.fragWhh[nt + 1][lane][0];
      }
      const int cb = nt & 1;
      v8f zc = {};
      acc[nt] = __builtin_amdgcn_wmma_f32_16x16x32_f16(
          false, ax2.v, false, bs[cb][0], (short)0, zc, false, false);
      acc[nt] = __builtin_amdgcn_wmma_f32_16x16x32_f16(
          false, ax0.v, false, bs[cb][1], (short)0, acc[nt], false, false);
      acc[nt] = __builtin_amdgcn_wmma_f32_16x16x32_f16(
          false, ax1.v, false, bs[cb][2], (short)0, acc[nt], false, false);
      acc[nt] = __builtin_amdgcn_wmma_f32_16x16x32_f16(
          false, hfrag.v, false, bs[cb][3], (short)0, acc[nt], false, false);
    }

    // ---- Activations + state update (tiles: 0-1=i, 2-3=f, 4-5=g, 6-7=o) ----
    U8f hn[2];
#pragma unroll
    for (int j = 0; j < 2; ++j) {
      U8f ig, fg, gg, og, cc;
      ig.v = acc[j];
      fg.v = acc[2 + j];
      gg.v = acc[4 + j];
      og.v = acc[6 + j];
      cc.v = (j == 0) ? c0 : c1;
#pragma unroll
      for (int r = 0; r < 8; ++r) {
        float i_ = sigmoid_f(ig.e[r]);
        float f_ = sigmoid_f(fg.e[r]);
        float g_ = tanh_fast(gg.e[r]);
        float o_ = sigmoid_f(og.e[r]);
        float c_ = f_ * cc.e[r] + i_ * g_;
        cc.e[r]  = c_;
        hn[j].e[r] = o_ * tanh_fast(c_);
      }
      if (j == 0) c0 = cc.v; else c1 = cc.v;
    }

    // ---- Transpose h (C-layout) -> A-fragment via per-wave LDS tile ----
#pragma unroll
    for (int j = 0; j < 2; ++j) {
#pragma unroll
      for (int r = 0; r < 8; ++r) {
        int row = r + 8 * lh;
        int col = 16 * j + lm;
        smem.hbuf[wave][row][col] = (_Float16)hn[j].e[r];
      }
    }
    __syncthreads();
    hfrag.h2[0] = *(const v8h*)&smem.hbuf[wave][lm][ko];
    hfrag.h2[1] = *(const v8h*)&smem.hbuf[wave][lm][ko + 16];
  }

  // ---- Final projection: out[b, o] = h_T[b,:] . Wout[o,:] + bout[o] ----
  float accp = bout[lh];
#pragma unroll
  for (int k = 0; k < HID; ++k)
    accp += (float)smem.hbuf[wave][lm][k] * Wout[lh * HID + k];
  out[(size_t)(rowBase + lm) * 2 + lh] = accp;
}

extern "C" void kernel_launch(void* const* d_in, const int* in_sizes, int n_in,
                              void* d_out, int out_size, void* d_ws, size_t ws_size,
                              hipStream_t stream) {
  (void)in_sizes; (void)n_in; (void)d_ws; (void)ws_size; (void)out_size;
  const float* x    = (const float*)d_in[0];
  const float* Wih  = (const float*)d_in[1];
  const float* Whh  = (const float*)d_in[2];
  const float* bih  = (const float*)d_in[3];
  const float* bhh  = (const float*)d_in[4];
  const float* Wout = (const float*)d_in[5];
  const float* bout = (const float*)d_in[6];
  float* out = (float*)d_out;

  lstm_fused_kernel<<<dim3(NBLOCKS), dim3(128), 0, stream>>>(
      x, Wih, Whh, bih, bhh, Wout, bout, out);
}
